// MoELayer_31275951850054
// MI455X (gfx1250) — compile-verified, matrix operand order checked by reading, not run on hardware
//
#include <hip/hip_runtime.h>
#include <hip/hip_bf16.h>

// MoE hash-router FFN for gfx1250 (MI455X).
//  - prep: W1/W2 fp32 -> bf16, pre-transposed for WMMA-B layout (once per launch)
//  - route: per-expert token gather lists (int atomics)
//  - ffn: per (expert, 64-token tile): fused bf16 WMMA two-GEMM with relu,
//         weight chunks staged into padded LDS via Tensor Data Mover,
//         scaled global_atomic_add_f32 scatter into out.

#define DIMX 512
#define HIDX 2048
#define NEXP 8
#define TM   64      // tokens per tile
#define HC   64      // hidden chunk width
#define SXK  520     // xs row stride (elems; pad 512->520 : 16B aligned, bank-spread)
#define SWK  520     // w1 col stride (K dim = 512, same padding)
#define SW2  72      // w2 col stride (K dim = 64 -> 72)
#define SH   72      // hs row stride (cols = 64 -> 72)

typedef __attribute__((ext_vector_type(16))) __bf16 v16bf;
typedef __attribute__((ext_vector_type(8)))  __bf16 v8bf;
typedef __attribute__((ext_vector_type(8)))  float  v8f;
typedef __attribute__((ext_vector_type(4)))  unsigned int uint32x4;
typedef __attribute__((ext_vector_type(8)))  int int32x8;
typedef __attribute__((ext_vector_type(4)))  int int32x4;

union V16 { v16bf v; v8bf h[2]; };

#if __has_builtin(__builtin_amdgcn_tensor_load_to_lds)
#define USE_TDM 1
#else
#define USE_TDM 0
#endif

__device__ __forceinline__ unsigned short f2bf(float f) {
    unsigned u = __float_as_uint(f);
    unsigned r = u + 0x7FFFu + ((u >> 16) & 1u);   // round-to-nearest-even
    return (unsigned short)(r >> 16);
}

#if USE_TDM
// 2-D tile load via Tensor Data Mover. Units: elements of 2 bytes (bf16).
// D# fields per CDNA5 ISA §8.3/§8.4; pad_* reproduce padded LDS row strides.
__device__ __forceinline__ void tdm_load_2d(unsigned lds_off, const unsigned short* gptr,
                                            unsigned tile_d0, unsigned tile_d1,
                                            unsigned stride_d0,
                                            unsigned pad_interval, unsigned pad_amount) {
    unsigned long long ga = (unsigned long long)(const void*)gptr;
    uint32x4 g0;
    g0[0] = 1u;                                              // count=1, user, no gather
    g0[1] = lds_off;                                         // lds_addr (bytes)
    g0[2] = (unsigned)(ga & 0xFFFFFFFFu);                    // global_addr[31:0]
    g0[3] = (unsigned)((ga >> 32) & 0x1FFFFFFu) | (2u << 30);// global_addr[56:32] | type=2
    int32x8 g1;
    g1[0] = (int)((1u << 16) | (1u << 20) |                  // data_size=2B, pad_enable
                  (pad_interval << 22) | (pad_amount << 25));
    g1[1] = (int)(tile_d0 << 16);                            // tensor_dim0[15:0] (== tile_d0)
    g1[2] = (int)((tile_d0 >> 16) | (tile_d1 << 16));        // tensor_dim0 hi | tensor_dim1 lo
    g1[3] = (int)((tile_d1 >> 16) | (tile_d0 << 16));        // tensor_dim1 hi | tile_dim0
    g1[4] = (int)(tile_d1);                                  // tile_dim1 (tile_dim2 = 0)
    g1[5] = (int)(stride_d0);                                // tensor_dim0_stride[31:0]
    g1[6] = 0;                                               // stride hi | dim1_stride lo
    g1[7] = 0;
    int32x4 z4 = {0, 0, 0, 0};
    int32x8 z8 = {0, 0, 0, 0, 0, 0, 0, 0};
    __builtin_amdgcn_tensor_load_to_lds(g0, g1, z4, z4, z8, 0);
}
#endif

__global__ void moe_zero_kernel(float* __restrict__ out, size_t n, int* __restrict__ counts) {
    size_t stride = (size_t)gridDim.x * blockDim.x;
    for (size_t i = (size_t)blockIdx.x * blockDim.x + threadIdx.x; i < n; i += stride)
        out[i] = 0.0f;
    if (blockIdx.x == 0 && threadIdx.x < NEXP) counts[threadIdx.x] = 0;
}

__global__ void moe_route_kernel(const int* __restrict__ assign, int* __restrict__ counts,
                                 int* __restrict__ tidx, float* __restrict__ twgt, int n_tok) {
    int t = blockIdx.x * blockDim.x + threadIdx.x;
    if (t >= n_tok) return;
    int a0 = assign[2 * t + 0];
    int a1 = assign[2 * t + 1];
    if (a0 == a1) {
        int p = atomicAdd(&counts[a0], 1);
        tidx[a0 * n_tok + p] = t; twgt[a0 * n_tok + p] = 1.0f;
    } else {
        int p0 = atomicAdd(&counts[a0], 1);
        tidx[a0 * n_tok + p0] = t; twgt[a0 * n_tok + p0] = 0.5f;
        int p1 = atomicAdd(&counts[a1], 1);
        tidx[a1 * n_tok + p1] = t; twgt[a1 * n_tok + p1] = 0.5f;
    }
}

// Transpose+convert: src [R][C] f32 (expert = blockIdx.z) -> dst [C][R] bf16.
__global__ __launch_bounds__(256)
void moe_tw_kernel(const float* __restrict__ src, unsigned short* __restrict__ dst, int R, int C) {
    __shared__ unsigned short tile[64][72];
    const float* s = src + (size_t)blockIdx.z * R * C;
    unsigned short* d = dst + (size_t)blockIdx.z * R * C;
    const int r0 = blockIdx.y * 64, c0 = blockIdx.x * 64;
    const int tid = threadIdx.x;
    #pragma unroll
    for (int it = 0; it < 4; ++it) {
        int id = it * 256 + tid;
        int r = id >> 4, c4 = (id & 15) << 2;
        float4 v = *(const float4*)(s + (size_t)(r0 + r) * C + c0 + c4);
        tile[r][c4 + 0] = f2bf(v.x); tile[r][c4 + 1] = f2bf(v.y);
        tile[r][c4 + 2] = f2bf(v.z); tile[r][c4 + 3] = f2bf(v.w);
    }
    __syncthreads();
    #pragma unroll
    for (int it = 0; it < 4; ++it) {
        int id = it * 256 + tid;
        int c = id >> 4, r4 = (id & 15) << 2;
        uint2 w;
        w.x = (unsigned)tile[r4 + 0][c] | ((unsigned)tile[r4 + 1][c] << 16);
        w.y = (unsigned)tile[r4 + 2][c] | ((unsigned)tile[r4 + 3][c] << 16);
        *(uint2*)(d + (size_t)(c0 + c) * R + r0 + r4) = w;
    }
}

__global__ __launch_bounds__(256, 1)
void moe_ffn_kernel(const float* __restrict__ x,
                    const unsigned short* __restrict__ W1t,   // [E][HID][D] bf16
                    const float* __restrict__ b1,
                    const unsigned short* __restrict__ W2t,   // [E][D][HID] bf16
                    const float* __restrict__ b2,
                    const int* __restrict__ counts,
                    const int* __restrict__ tidx, const float* __restrict__ twgt,
                    float* __restrict__ out, int n_tok) {
    const int e    = blockIdx.y;
    const int cnt  = counts[e];
    const int base = blockIdx.x * TM;
    if (base >= cnt) return;

    __shared__ unsigned short xs[TM * SXK];        //  66,560 B : bf16 token rows
    __shared__ unsigned short wbuf[DIMX * SW2];    //  73,728 B : W1 chunk / W2 chunk (time-muxed)
    __shared__ unsigned short hs[TM * SH];         //   9,216 B : bf16 relu(h) chunk
    __shared__ int   tok_s[TM];
    __shared__ float tw_s[TM];

    const int tid  = threadIdx.x;
    const int lane = tid & 31;
    const int wave = tid >> 5;
    const int lmod = lane & 15;
    const int c8   = (lane >> 4) * 8;              // K-half select per WMMA 16-bit layout

    if (tid < TM) {
        int g = base + tid;
        if (g < cnt) { tok_s[tid] = tidx[e * n_tok + g]; tw_s[tid] = twgt[e * n_tok + g]; }
        else         { tok_s[tid] = 0;                   tw_s[tid] = 0.0f; }
    }
    __syncthreads();

    // Stage x tile fp32 -> bf16 (64 x 512), float4-coalesced (gathered rows).
    for (int it = 0; it < 32; ++it) {
        int id = it * 256 + tid;
        int r  = id >> 7;
        int c4 = (id & 127) << 2;
        const float4 v = *(const float4*)(x + (size_t)tok_s[r] * DIMX + c4);
        unsigned short* p = &xs[r * SXK + c4];
        p[0] = f2bf(v.x); p[1] = f2bf(v.y); p[2] = f2bf(v.z); p[3] = f2bf(v.w);
    }

    const unsigned short* W1e = W1t + (size_t)e * DIMX * HIDX;  // [HID][D] bf16
    const unsigned short* W2e = W2t + (size_t)e * DIMX * HIDX;  // [D][HID] bf16
#if USE_TDM
    const unsigned wbuf_lds = (unsigned)(unsigned long long)(const void*)&wbuf[0];
#endif

    v8f acc2[16];
    #pragma unroll
    for (int t = 0; t < 16; ++t) acc2[t] = (v8f){};

    for (int hc = 0; hc < HIDX; hc += HC) {
        __syncthreads();   // previous chunk done reading wbuf / hs

        // Stage W1t rows hc..hc+63 ([64][512] bf16) into wbuf[c*SWK + k].
#if USE_TDM
        if (wave == 0) {
            // 64 rows of 512 elems; LDS pad 16B every 1024B (256 dw -> code 7, 4 dw -> code 3)
            tdm_load_2d(wbuf_lds, W1e + (size_t)hc * DIMX, DIMX, HC, DIMX, 7u, 3u);
            __builtin_amdgcn_s_wait_tensorcnt(0);
        }
#else
        for (int it = 0; it < 16; ++it) {
            int id = it * 256 + tid;            // 4096 x 16B
            int c = id >> 6, koff = (id & 63) << 3;
            v8bf v = *(const v8bf*)(W1e + (size_t)(hc + c) * DIMX + koff);
            *(v8bf*)&wbuf[c * SWK + koff] = v;
        }
#endif
        __syncthreads();

        // GEMM1: h(64x64) = relu(xs(64x512) @ W1chunk(512x64) + b1).
        // Wave owns col-tile ct = wave&3, row-tiles rt0 and rt0+2; B feeds 2 wmma/k-step.
        {
            const int rt0 = wave >> 2, ct = wave & 3;
            v8f accA = (v8f){};
            v8f accB = (v8f){};
            #pragma unroll
            for (int ks = 0; ks < 16; ++ks) {
                const int k0 = ks * 32;
                V16 a0, a1, b;
                a0.h[0] = *(const v8bf*)&xs[(rt0 * 16 + lmod) * SXK + k0 + c8];
                a0.h[1] = *(const v8bf*)&xs[(rt0 * 16 + lmod) * SXK + k0 + c8 + 16];
                a1.h[0] = *(const v8bf*)&xs[((rt0 + 2) * 16 + lmod) * SXK + k0 + c8];
                a1.h[1] = *(const v8bf*)&xs[((rt0 + 2) * 16 + lmod) * SXK + k0 + c8 + 16];
                b.h[0]  = *(const v8bf*)&wbuf[(ct * 16 + lmod) * SWK + k0 + c8];
                b.h[1]  = *(const v8bf*)&wbuf[(ct * 16 + lmod) * SWK + k0 + c8 + 16];
                accA = __builtin_amdgcn_wmma_f32_16x16x32_bf16(
                         false, a0.v, false, b.v, (short)0, accA, false, false);
                accB = __builtin_amdgcn_wmma_f32_16x16x32_bf16(
                         false, a1.v, false, b.v, (short)0, accB, false, false);
            }
            const float b1v = b1[(size_t)e * HIDX + hc + ct * 16 + lmod];
            #pragma unroll
            for (int j = 0; j < 8; ++j) {
                int m0 = rt0 * 16 + (lane >> 4) * 8 + j;        // C-layout: M = j + 8*(lane/16)
                int m1 = (rt0 + 2) * 16 + (lane >> 4) * 8 + j;
                hs[m0 * SH + ct * 16 + lmod] = f2bf(fmaxf(accA[j] + b1v, 0.0f));
                hs[m1 * SH + ct * 16 + lmod] = f2bf(fmaxf(accB[j] + b1v, 0.0f));
            }
        }
        __syncthreads();   // hs complete, GEMM1 done reading wbuf

        // L2 prefetch of next chunk's weights.
        if (hc + HC < HIDX) {
            __builtin_prefetch(W1e + (size_t)(hc + HC + (tid >> 2)) * DIMX + (tid & 3) * 128, 0, 1);
            __builtin_prefetch(W2e + (size_t)(tid * 2) * HIDX + hc + HC, 0, 1);
        }

        // Stage W2t column-chunk ([512 rows][64 elems @ hc]) into wbuf[n*SW2 + k].
#if USE_TDM
        if (wave == 0) {
            // 512 rows of 64 elems, memory row stride HID; LDS pad 16B every 128B
            tdm_load_2d(wbuf_lds, W2e + hc, HC, DIMX, HIDX, 4u, 3u);
            __builtin_amdgcn_s_wait_tensorcnt(0);
        }
#else
        for (int it = 0; it < 16; ++it) {
            int id = it * 256 + tid;            // 4096 x 16B
            int n = id >> 3, koff = (id & 7) << 3;
            v8bf v = *(const v8bf*)(W2e + (size_t)n * HIDX + hc + koff);
            *(v8bf*)&wbuf[n * SW2 + koff] = v;
        }
#endif
        __syncthreads();

        // GEMM2: y(64x512) += hs(64x64) @ W2chunk(64x512); wave owns 64 output cols,
        // 4 row-tiles x 4 col-tiles of persistent f32 accumulators.
        {
            const int n0 = wave * 64;
            #pragma unroll
            for (int ks = 0; ks < 2; ++ks) {
                const int k0 = ks * 32;
                V16 a[4];
                #pragma unroll
                for (int rt = 0; rt < 4; ++rt) {
                    a[rt].h[0] = *(const v8bf*)&hs[(rt * 16 + lmod) * SH + k0 + c8];
                    a[rt].h[1] = *(const v8bf*)&hs[(rt * 16 + lmod) * SH + k0 + c8 + 16];
                }
                #pragma unroll
                for (int ct = 0; ct < 4; ++ct) {
                    V16 b;
                    b.h[0] = *(const v8bf*)&wbuf[(n0 + ct * 16 + lmod) * SW2 + k0 + c8];
                    b.h[1] = *(const v8bf*)&wbuf[(n0 + ct * 16 + lmod) * SW2 + k0 + c8 + 16];
                    #pragma unroll
                    for (int rt = 0; rt < 4; ++rt) {
                        acc2[rt * 4 + ct] = __builtin_amdgcn_wmma_f32_16x16x32_bf16(
                            false, a[rt].v, false, b.v, (short)0, acc2[rt * 4 + ct], false, false);
                    }
                }
            }
        }
    }

    // Epilogue: +b2, scale by routing weight, atomic scatter into out.
    {
        const int n0 = wave * 64;
        #pragma unroll
        for (int rt2 = 0; rt2 < 4; ++rt2) {
            #pragma unroll
            for (int ct = 0; ct < 4; ++ct) {
                const int n = n0 + ct * 16 + lmod;
                const float b2v = b2[(size_t)e * DIMX + n];
                v8f a = acc2[rt2 * 4 + ct];
                #pragma unroll
                for (int j = 0; j < 8; ++j) {
                    int m = rt2 * 16 + (lane >> 4) * 8 + j;
                    float v = (a[j] + b2v) * tw_s[m];
                    unsafeAtomicAdd(&out[(size_t)tok_s[m] * DIMX + n], v);
                }
            }
        }
    }
}

extern "C" void kernel_launch(void* const* d_in, const int* in_sizes, int n_in,
                              void* d_out, int out_size, void* d_ws, size_t ws_size,
                              hipStream_t stream) {
    const float* x      = (const float*)d_in[0];
    const int*   assign = (const int*)d_in[1];
    const float* W1     = (const float*)d_in[2];
    const float* b1     = (const float*)d_in[3];
    const float* W2     = (const float*)d_in[4];
    const float* b2     = (const float*)d_in[5];
    float* out = (float*)d_out;

    const int n_tok = in_sizes[0] / DIMX;   // B*S = 32768

    char* wsb = (char*)d_ws;
    const size_t list_bytes = (size_t)NEXP * n_tok * 4;
    int*   counts = (int*)wsb;                                   // 8 ints
    int*   tidx   = (int*)(wsb + 256);
    float* twgt   = (float*)(wsb + 256 + list_bytes);
    unsigned short* W1t = (unsigned short*)(wsb + 256 + 2 * list_bytes);                    // 16 MB
    unsigned short* W2t = (unsigned short*)(wsb + 256 + 2 * list_bytes +
                                            (size_t)NEXP * DIMX * HIDX * 2);                // 16 MB

    moe_zero_kernel<<<4096, 256, 0, stream>>>(out, (size_t)out_size, counts);
    moe_route_kernel<<<(n_tok + 255) / 256, 256, 0, stream>>>(assign, counts, tidx, twgt, n_tok);

    // W1 [E][512][2048] f32 -> W1t [E][2048][512] bf16
    moe_tw_kernel<<<dim3(HIDX / 64, DIMX / 64, NEXP), 256, 0, stream>>>(W1, W1t, DIMX, HIDX);
    // W2 [E][2048][512] f32 -> W2t [E][512][2048] bf16
    moe_tw_kernel<<<dim3(DIMX / 64, HIDX / 64, NEXP), 256, 0, stream>>>(W2, W2t, HIDX, DIMX);

    dim3 grid(n_tok / TM, NEXP);            // blocks past a list's end exit immediately
    moe_ffn_kernel<<<grid, 256, 0, stream>>>(x, W1t, b1, W2t, b2, counts, tidx, twgt, out, n_tok);
}
